// Int8BaichuanMLP_49460843380755
// MI455X (gfx1250) — 8-bit matrix operand order repaired and checked by reading, compiled, not run on hardware
//
#include <hip/hip_runtime.h>
#include <hip/hip_bf16.h>

#define HDIM 4096
#define IDIM 11008
#define MT   16

typedef __attribute__((ext_vector_type(8))) int v8i;

union AFrag { v8i v; long long d[4]; };
union BFrag { v8i v; int4 q[2]; };

__global__ __launch_bounds__(256) void
int8_baichuan_mlp_kernel(const signed char* __restrict__ x,
                         const signed char* __restrict__ wg,
                         const signed char* __restrict__ wu,
                         const signed char* __restrict__ wd,
                         const float* __restrict__ bg,
                         const float* __restrict__ bu,
                         const float* __restrict__ bd,
                         const float* __restrict__ p_ag,
                         const float* __restrict__ p_au,
                         const float* __restrict__ p_dqs,
                         const float* __restrict__ p_ad,
                         float* __restrict__ out)
{
    extern __shared__ char smem[];
    signed char* qbuf  = (signed char*)smem;
    signed char* xtile = (signed char*)(smem + MT * IDIM);

    const size_t rowBase = (size_t)blockIdx.x * MT;

    {
        const signed char* src = x + rowBase * HDIM;
        const unsigned ldsBase = (unsigned)(MT * IDIM);
        #pragma unroll
        for (int it = 0; it < (MT * HDIM) / (256 * 16); ++it) {
            const unsigned off = (unsigned)((it * 256 + (int)threadIdx.x) * 16);
            asm volatile("global_load_async_to_lds_b128 %0, %1, off"
                         :: "v"(ldsBase + off), "v"(src + off)
                         : "memory");
        }
        asm volatile("s_wait_asynccnt 0x0" ::: "memory");
    }
    __syncthreads();

    const int lane = threadIdx.x & 31;
    const int wave = threadIdx.x >> 5;
    const int kh   = lane >> 4;
    const int ln   = lane & 15;

    const float ag      = *p_ag;
    const float au      = *p_au;
    const float inv_dqs = 1.0f / *p_dqs;
    const float ad      = *p_ad;

    {
        const int tilesI   = IDIM / 16;
        const int perWave  = tilesI / 8;
        const signed char* __restrict__ xrow = xtile + ln * HDIM;
        for (int t = wave * perWave; t < (wave + 1) * perWave; ++t) {
            const int colBase = t * 16;
            const signed char* __restrict__ wgrow = wg + (size_t)(colBase + ln) * HDIM;
            const signed char* __restrict__ wurow = wu + (size_t)(colBase + ln) * HDIM;
            v8i accG = {};
            v8i accU = {};
            #pragma unroll 4
            for (int k = 0; k < HDIM; k += 64) {
                AFrag a;
                a.d[0] = *(const long long*)(xrow + k +  0 + 16 * kh);
                a.d[1] = *(const long long*)(xrow + k +  8 + 16 * kh);
                a.d[2] = *(const long long*)(xrow + k + 32 + 16 * kh);
                a.d[3] = *(const long long*)(xrow + k + 40 + 16 * kh);
                BFrag bG, bU;
                bG.q[0] = *(const int4*)(wgrow + k +  0 + 16 * kh);
                bG.q[1] = *(const int4*)(wgrow + k + 32 + 16 * kh);
                bU.q[0] = *(const int4*)(wurow + k +  0 + 16 * kh);
                bU.q[1] = *(const int4*)(wurow + k + 32 + 16 * kh);
                accG = __builtin_amdgcn_wmma_i32_16x16x64_iu8(true, a.v, true, bG.v, accG, false, false);
                accU = __builtin_amdgcn_wmma_i32_16x16x64_iu8(true, a.v, true, bU.v, accU, false, false);
            }
            const float bgv = bg[colBase + ln];
            const float buv = bu[colBase + ln];
            #pragma unroll
            for (int r = 0; r < 8; ++r) {
                float g = (float)accG[r] * ag + bgv;
                float u = (float)accU[r] * au + buv;
                _Float16 gh = (_Float16)g;
                float gf = (float)gh;
                float sil = gf / (1.0f + __expf(-gf));
                float h = sil * u;
                int qi = (int)rintf(h * inv_dqs);
                qi = qi < -128 ? -128 : (qi > 127 ? 127 : qi);
                const int m = r + 8 * kh;
                qbuf[m * IDIM + colBase + ln] = (signed char)qi;
            }
        }
    }
    __syncthreads();

    {
        const int tilesH  = HDIM / 16;
        const int perWave = tilesH / 8;
        const signed char* __restrict__ qrow = qbuf + ln * IDIM;
        for (int t = wave * perWave; t < (wave + 1) * perWave; ++t) {
            const int colBase = t * 16;
            const signed char* __restrict__ wdrow = wd + (size_t)(colBase + ln) * IDIM;
            v8i acc = {};
            #pragma unroll 4
            for (int k = 0; k < IDIM; k += 64) {
                AFrag a;
                a.d[0] = *(const long long*)(qrow + k +  0 + 16 * kh);
                a.d[1] = *(const long long*)(qrow + k +  8 + 16 * kh);
                a.d[2] = *(const long long*)(qrow + k + 32 + 16 * kh);
                a.d[3] = *(const long long*)(qrow + k + 40 + 16 * kh);
                BFrag b;
                b.q[0] = *(const int4*)(wdrow + k +  0 + 16 * kh);
                b.q[1] = *(const int4*)(wdrow + k + 32 + 16 * kh);
                acc = __builtin_amdgcn_wmma_i32_16x16x64_iu8(true, a.v, true, b.v, acc, false, false);
            }
            const float bdv = bd[colBase + ln];
            #pragma unroll
            for (int r = 0; r < 8; ++r) {
                const int m = r + 8 * kh;
                out[(rowBase + m) * (size_t)HDIM + colBase + ln] = (float)acc[r] * ad + bdv;
            }
        }
    }
}

extern "C" void kernel_launch(void* const* d_in, const int* in_sizes, int n_in,
                              void* d_out, int out_size, void* d_ws, size_t ws_size,
                              hipStream_t stream) {
    (void)n_in; (void)out_size; (void)d_ws; (void)ws_size;
    const signed char* x  = (const signed char*)d_in[0];
    const signed char* wg = (const signed char*)d_in[1];
    const signed char* wu = (const signed char*)d_in[2];
    const signed char* wd = (const signed char*)d_in[3];
    const float* bg  = (const float*)d_in[4];
    const float* bu  = (const float*)d_in[5];
    const float* bd  = (const float*)d_in[6];
    const float* ag  = (const float*)d_in[7];
    const float* au  = (const float*)d_in[8];
    const float* dqs = (const float*)d_in[9];
    const float* ad  = (const float*)d_in[10];
    float* out = (float*)d_out;

    const int M = in_sizes[0] / HDIM;
    const size_t smemBytes = (size_t)MT * IDIM + (size_t)MT * HDIM;

    hipFuncSetAttribute((const void*)int8_baichuan_mlp_kernel,
                        hipFuncAttributeMaxDynamicSharedMemorySize, (int)smemBytes);

    int8_baichuan_mlp_kernel<<<dim3(M / MT), dim3(256), smemBytes, stream>>>(
        x, wg, wu, wd, bg, bu, bd, ag, au, dqs, ad, out);
}
